// EncoderRNN_42228118454326
// MI455X (gfx1250) — matrix operand order repaired and checked by reading, not compiled, not run on hardware
//
#include <hip/hip_runtime.h>
#include <hip/hip_bf16.h>

// EncoderRNN (2-layer LSTM, batch=1) for MI455X / gfx1250.
//
// Roofline reasoning:
//  * Input projections (T x H) @ (H x 4H) per layer = 34 GFLOP each, fully
//    parallel over T -> bf16 WMMA (v_wmma_f32_16x16x32_bf16), f32 accum.
//    Each wave owns a 16(M) x 64(N) strip: the A fragment is loaded once per
//    K-step and reused for 4 WMMAs (register reuse, 4x arithmetic intensity
//    vs one-tile-per-wave).
//  * Recurrent matvec W_hh @ h is batch-1 (N=1): WMMA would be 1/16 utilized,
//    so it runs as fp32 VALU dot products. W_hh (16.8 MB/layer) is L2-resident
//    (192 MB L2); a persistent 64-block kernel with a per-step grid barrier
//    spreads the per-step L2 traffic across WGPs while honoring the t->t+1
//    dependency. Cell state c stays in registers across all 4096 steps.

#define HDIM 1024
#define NL   2
#define TSEQ 4096
#define G4H  4096   // 4*HDIM
#define RBLOCKS 64

typedef __attribute__((ext_vector_type(16))) __bf16 v16bf;
typedef __attribute__((ext_vector_type(8)))  __bf16 v8bf;
typedef __attribute__((ext_vector_type(8)))  float  v8f;

union FragBF { v16bf v; v8bf h[2]; };

__device__ __forceinline__ v8bf ld_v8bf(const __hip_bfloat16* p) {
  return *reinterpret_cast<const v8bf*>(p);
}

// ---------------------------------------------------------------------------
// Embedding gather, fp32 -> bf16 (one block per token, 256 threads x float4)
// ---------------------------------------------------------------------------
__global__ __launch_bounds__(256) void embed_gather_bf16(
    const int* __restrict__ tok, const float* __restrict__ emb,
    __hip_bfloat16* __restrict__ xbf) {
  const int t = blockIdx.x;
  const int row = tok[t];
  const float4 v =
      reinterpret_cast<const float4*>(emb + (size_t)row * HDIM)[threadIdx.x];
  __hip_bfloat16* dst = xbf + (size_t)t * HDIM + threadIdx.x * 4;
  dst[0] = __float2bfloat16(v.x);
  dst[1] = __float2bfloat16(v.y);
  dst[2] = __float2bfloat16(v.z);
  dst[3] = __float2bfloat16(v.w);
}

// ---------------------------------------------------------------------------
// fp32 -> bf16 (vectorized), used to down-convert W_ih per layer
// ---------------------------------------------------------------------------
__global__ __launch_bounds__(256) void f32_to_bf16_vec(
    const float* __restrict__ src, __hip_bfloat16* __restrict__ dst, int n4) {
  const int i = blockIdx.x * blockDim.x + threadIdx.x;
  if (i >= n4) return;
  const float4 v = reinterpret_cast<const float4*>(src)[i];
  __hip_bfloat16* d = dst + (size_t)i * 4;
  d[0] = __float2bfloat16(v.x);
  d[1] = __float2bfloat16(v.y);
  d[2] = __float2bfloat16(v.z);
  d[3] = __float2bfloat16(v.w);
}

// ---------------------------------------------------------------------------
// G[t, n] = sum_k X[t,k] * W[n,k] + (b_ih[n] + b_hh[n])
// bf16 WMMA 16x16x32, f32 accumulation. One wave per 16(M) x 64(N) strip
// (4 accumulators, A fragment reused 4x per K-step); block = 8 waves
// covering 16(M) x 512(N).
//
// A fragment (16x32, MxK): lane(half,m) holds K in {half*8..+7, 16+half*8..+7}
// B fragment (32x16, KxN): lane(half,n) holds K in {half*8..+7, 16+half*8..+7};
//   since B[k][n] = W[n][k], each lane reads two runs of 8 contiguous bf16 of row n of W.
// ---------------------------------------------------------------------------
__global__ __launch_bounds__(256) void gates_gemm_wmma(
    const __hip_bfloat16* __restrict__ X,   // [T, H] bf16, row-major
    const __hip_bfloat16* __restrict__ W,   // [4H, H] bf16, row-major
    const float* __restrict__ bih, const float* __restrict__ bhh,
    float* __restrict__ G)                  // [T, 4H] fp32
{
  const int lane = threadIdx.x & 31;
  const int wave = threadIdx.x >> 5;   // 0..7
  const int half = lane >> 4;
  const int mrow = lane & 15;

  const int n0 = (blockIdx.x * 8 + wave) * 64;   // 4 x 16 N-tiles per wave
  const int m0 = blockIdx.y * 16;

  const __hip_bfloat16* arow = X + (size_t)(m0 + mrow) * HDIM + half * 8;
  const __hip_bfloat16* brow = W + (size_t)(n0 + mrow) * HDIM + half * 8;

  v8f acc[4] = {v8f{}, v8f{}, v8f{}, v8f{}};

  for (int k0 = 0; k0 < HDIM; k0 += 32) {
    FragBF a;
    a.h[0] = ld_v8bf(arow + k0);        // K = k0 + half*8 .. +7
    a.h[1] = ld_v8bf(arow + k0 + 16);   // K = k0 + 16 + half*8 .. +7

    FragBF b[4];
#pragma unroll
    for (int nt = 0; nt < 4; ++nt) {
      const __hip_bfloat16* bp = brow + (size_t)nt * 16 * HDIM;
      b[nt].h[0] = ld_v8bf(bp + k0);      // K = k0 + half*8 .. +7
      b[nt].h[1] = ld_v8bf(bp + k0 + 16); // K = k0 + 16 + half*8 .. +7
    }
    if (k0 + 32 < HDIM) {
      __builtin_prefetch(arow + k0 + 32, 0, 3);   // global_prefetch
      __builtin_prefetch(brow + k0 + 32, 0, 3);
    }
#pragma unroll
    for (int nt = 0; nt < 4; ++nt) {
      acc[nt] = __builtin_amdgcn_wmma_f32_16x16x32_bf16(
          /*neg_a=*/false, a.v, /*neg_b=*/false, b[nt].v,
          /*c_mod=*/(short)0, acc[nt], /*reuse_a=*/false, /*reuse_b=*/false);
    }
  }

  // C/D layout: VGPR r, lane -> (row = r + 8*half, col = lane%16)
#pragma unroll
  for (int nt = 0; nt < 4; ++nt) {
    const int col = n0 + nt * 16 + mrow;
    const float bias = bih[col] + bhh[col];
    float* gout = G + (size_t)(m0 + 8 * half) * G4H + col;
#pragma unroll
    for (int r = 0; r < 8; ++r) gout[(size_t)r * G4H] = acc[nt][r] + bias;
  }
}

// ---------------------------------------------------------------------------
// Persistent recurrence: 64 blocks x 256 threads = 512 waves, 2 hidden units
// per wave. Per step: 8 fp32 dot products of length H (wave-parallel, shuffle
// reduced), gate nonlinearities in lane 0, double-buffered h, grid barrier.
// ---------------------------------------------------------------------------
__global__ __launch_bounds__(256) void lstm_layer_rec(
    const float* __restrict__ Whh,            // [4H, H] fp32
    const float* __restrict__ G,              // [T, 4H] fp32 (precomputed)
    float* __restrict__ hbuf,                 // [2*H] fp32 ping/pong (zeroed)
    float* __restrict__ outs_f32,             // [T, H] or nullptr
    __hip_bfloat16* __restrict__ outs_bf16,   // [T, H] or nullptr
    float* __restrict__ hT, float* __restrict__ cT,
    int* __restrict__ bar)                    // [2]: count, sense (zeroed)
{
  const int lane = threadIdx.x & 31;
  const int wave = blockIdx.x * (blockDim.x >> 5) + (threadIdx.x >> 5);
  const int j0 = wave * 2;
  const int nblk = gridDim.x;

  __shared__ int s_phase;
  if (threadIdx.x == 0) s_phase = 0;
  __syncthreads();

  float c[2] = {0.f, 0.f};

  for (int t = 0; t < TSEQ; ++t) {
    const float4* h4 =
        reinterpret_cast<const float4*>(hbuf + ((t & 1) ? HDIM : 0));
    float* hw = hbuf + ((t & 1) ? 0 : HDIM);
    const float* Gr = G + (size_t)t * G4H;

#pragma unroll
    for (int jj = 0; jj < 2; ++jj) {
      const int j = j0 + jj;
      float dot[4];
#pragma unroll
      for (int g = 0; g < 4; ++g) {
        const float4* w4 = reinterpret_cast<const float4*>(
            Whh + (size_t)(j + g * HDIM) * HDIM);
        float s = 0.f;
#pragma unroll
        for (int i = 0; i < 8; ++i) {
          const float4 a = w4[lane + 32 * i];
          const float4 b = h4[lane + 32 * i];
          s = fmaf(a.x, b.x, s); s = fmaf(a.y, b.y, s);
          s = fmaf(a.z, b.z, s); s = fmaf(a.w, b.w, s);
        }
#pragma unroll
        for (int off = 16; off > 0; off >>= 1) s += __shfl_down(s, off, 32);
        dot[g] = s;
      }
      if (lane == 0) {
        const float ig = Gr[j]            + dot[0];
        const float fg = Gr[j + HDIM]     + dot[1];
        const float gg = Gr[j + 2 * HDIM] + dot[2];
        const float og = Gr[j + 3 * HDIM] + dot[3];
        const float is = 1.f / (1.f + __expf(-ig));
        const float fs = 1.f / (1.f + __expf(-fg));
        const float gt = tanhf(gg);
        const float os = 1.f / (1.f + __expf(-og));
        c[jj] = fs * c[jj] + is * gt;
        const float hv = os * tanhf(c[jj]);
        hw[j] = hv;
        if (outs_f32)  outs_f32[(size_t)t * HDIM + j] = hv;
        if (outs_bf16) outs_bf16[(size_t)t * HDIM + j] = __float2bfloat16(hv);
        if (t == TSEQ - 1) { hT[j] = hv; cT[j] = c[jj]; }
      }
    }

    // ---- grid-wide barrier (sense-reversing; all blocks co-resident) ----
    __threadfence();
    __syncthreads();
    if (threadIdx.x == 0) {
      const int p = s_phase;
      if (atomicAdd(&bar[0], 1) == nblk - 1) {
        atomicExch(&bar[0], 0);
        __threadfence();
        atomicExch(&bar[1], p ^ 1);
      } else {
        while (atomicAdd(&bar[1], 0) == p) { __builtin_amdgcn_s_sleep(1); }
      }
      s_phase = p ^ 1;
    }
    __syncthreads();
    __threadfence();
  }
}

// ---------------------------------------------------------------------------
extern "C" void kernel_launch(void* const* d_in, const int* in_sizes, int n_in,
                              void* d_out, int out_size, void* d_ws,
                              size_t ws_size, hipStream_t stream) {
  (void)in_sizes; (void)n_in; (void)out_size; (void)ws_size;
  const int*   tok = (const int*)  d_in[0];
  const float* emb = (const float*)d_in[1];
  const float* Wih = (const float*)d_in[2];   // [L, 4H, H]
  const float* Whh = (const float*)d_in[3];   // [L, 4H, H]
  const float* bih = (const float*)d_in[4];   // [L, 4H]
  const float* bhh = (const float*)d_in[5];   // [L, 4H]

  float* out  = (float*)d_out;
  float* outs = out;                              // [T, H]
  float* hT   = out + (size_t)TSEQ * HDIM;        // [L, 1, H]
  float* cT   = hT + (size_t)NL * HDIM;           // [L, 1, H]

  // Workspace layout: Xbf (8 MiB) | Wbf (8 MiB) | G (64 MiB) | hbuf | bar
  char* ws = (char*)d_ws;
  __hip_bfloat16* Xbf = (__hip_bfloat16*)ws;
  __hip_bfloat16* Wbf = (__hip_bfloat16*)(ws + (size_t)TSEQ * HDIM * 2);
  float* G    = (float*)(ws + (size_t)TSEQ * HDIM * 2 + (size_t)G4H * HDIM * 2);
  float* hbuf = (float*)((char*)G + (size_t)TSEQ * G4H * 4);
  int*   bar  = (int*)((char*)hbuf + 2 * HDIM * sizeof(float));

  const int n4w = (G4H * HDIM) / 4;
  const dim3 gemmGrid(G4H / (64 * 8), TSEQ / 16);   // waves own 16x64 strips

  for (int l = 0; l < NL; ++l) {
    // zero h ping/pong + barrier state (hipMemsetAsync is graph-capturable)
    hipMemsetAsync(hbuf, 0, 2 * HDIM * sizeof(float) + 2 * sizeof(int), stream);

    if (l == 0)
      embed_gather_bf16<<<TSEQ, 256, 0, stream>>>(tok, emb, Xbf);

    f32_to_bf16_vec<<<(n4w + 255) / 256, 256, 0, stream>>>(
        Wih + (size_t)l * G4H * HDIM, Wbf, n4w);

    gates_gemm_wmma<<<gemmGrid, 256, 0, stream>>>(
        Xbf, Wbf, bih + (size_t)l * G4H, bhh + (size_t)l * G4H, G);

    if (l == 0) {
      // layer-0 h-sequence feeds layer-1 GEMM: write bf16 straight into Xbf
      lstm_layer_rec<<<RBLOCKS, 256, 0, stream>>>(
          Whh + (size_t)l * G4H * HDIM, G, hbuf,
          nullptr, Xbf, hT + (size_t)l * HDIM, cT + (size_t)l * HDIM, bar);
    } else {
      lstm_layer_rec<<<RBLOCKS, 256, 0, stream>>>(
          Whh + (size_t)l * G4H * HDIM, G, hbuf,
          outs, nullptr, hT + (size_t)l * HDIM, cT + (size_t)l * HDIM, bar);
    }
  }
}
